// NeuralDictionaryV4_19430432047760
// MI455X (gfx1250) — matrix operand order repaired and checked by reading, not compiled, not run on hardware
//
#include <hip/hip_runtime.h>

// ---------- types ----------
typedef _Float16 h2  __attribute__((ext_vector_type(2)));
typedef _Float16 h8  __attribute__((ext_vector_type(8)));
typedef _Float16 v16h __attribute__((ext_vector_type(16)));
typedef float    v8f __attribute__((ext_vector_type(8)));
typedef unsigned int u32x4 __attribute__((ext_vector_type(4)));

#define B_Q   1024
#define N_K   32768
#define D_DIM 256
#define V_DIM 128

#define NSPLIT 32
#define KEYS_PER_WG (N_K / NSPLIT)   // 1024
#define CHUNKS (KEYS_PER_WG / 32)    // 32
#define WAVES 4
#define QTILES (B_Q / 16)            // 64
#define PART_STRIDE 2080             // 16*128 acc + 16 m + 16 l (floats)

#define L2E 1.4426950408889634f

// ---------- helpers ----------
static __device__ __forceinline__ h8 habs8(h8 x) {
  u32x4 u = __builtin_bit_cast(u32x4, x);
  u = u & (u32x4)0x7FFF7FFFu;
  return __builtin_bit_cast(h8, u);
}

static __device__ __forceinline__ float rsum8(h8 a) {
  return (float)a[0] + (float)a[1] + (float)a[2] + (float)a[3] +
         (float)a[4] + (float)a[5] + (float)a[6] + (float)a[7];
}

// swap with lane^16 (SWAPX16 ds_swizzle: and=0x1f, or=0, xor=0x10)
static __device__ __forceinline__ float xor16f(float x) {
  int r = __builtin_amdgcn_ds_swizzle(__builtin_bit_cast(int, x), 0x401f);
  return __builtin_bit_cast(float, r);
}

static __device__ __forceinline__ void dswait() {
  asm volatile("s_wait_dscnt 0" ::: "memory");
}

// ---------- kernel 1a: cast Q and K to f16 ----------
__global__ __launch_bounds__(256) void nd_cvt_qk(const float* __restrict__ q,
                                                 const float* __restrict__ k,
                                                 _Float16* __restrict__ Qh,
                                                 _Float16* __restrict__ Kh) {
  size_t i = (size_t)blockIdx.x * 256 + threadIdx.x;  // grid covers N_K*D_DIM
  if (i < (size_t)B_Q * D_DIM) Qh[i] = (_Float16)q[i];
  if (i < (size_t)N_K * D_DIM) Kh[i] = (_Float16)k[i];
}

// ---------- kernel 1b: tiled transpose of values -> f16 Vt[v][n] ----------
__global__ __launch_bounds__(256) void nd_transpose_v(const float* __restrict__ v,
                                                      _Float16* __restrict__ Vt) {
  __shared__ float tile[32][33];
  int bx = blockIdx.x;           // n tile: 0..N/32-1
  int by = blockIdx.y;           // v tile: 0..V/32-1
  int tx = threadIdx.x;          // 0..31
  int ty = threadIdx.y;          // 0..7
#pragma unroll
  for (int r = 0; r < 32; r += 8)
    tile[ty + r][tx] = v[(size_t)(bx * 32 + ty + r) * V_DIM + by * 32 + tx];
  __syncthreads();
#pragma unroll
  for (int r = 0; r < 32; r += 8)
    Vt[(size_t)(by * 32 + ty + r) * N_K + bx * 32 + tx] = (_Float16)tile[tx][ty + r];
}

// ---------- shared memory ----------
struct SMem {
  h8    qs[WAVES][16][32];   // 4 waves * 16 queries * 256 dims f16   = 32768 B
  uint4 vt4[128][4];         // V^T chunk: 128 v-dims x 32 keys f16   =  8192 B
  float S[WAVES][16][32];    // per-wave score tile f32               =  8192 B
  float scales[WAVES][16];   // per-wave softmax rescale broadcast    =   256 B
};                           // total 49408 B

// ---------- kernel 2: main flash-L1-attention ----------
__global__ __launch_bounds__(128) void nd_main(const _Float16* __restrict__ Qh,
                                               const _Float16* __restrict__ Kh,
                                               const _Float16* __restrict__ Vt,
                                               float* __restrict__ part) {
  __shared__ SMem sm;
  const int tid  = threadIdx.x;
  const int lane = tid & 31;
  const int w    = tid >> 5;
  const int qt   = blockIdx.x * WAVES + w;   // query tile 0..63
  const int split = blockIdx.y;              // 0..31
  const int n0   = split * KEYS_PER_WG;

  // ---- cooperative Q tile load: 64 queries x 256 dims f16 (32KB) ----
  {
    int ql = tid >> 1, hf = tid & 1;
    const uint4* src =
        (const uint4*)(Qh + (size_t)(blockIdx.x * 64 + ql) * D_DIM) + hf * 16;
    uint4* dst = (uint4*)&sm.qs[ql >> 4][ql & 15][hf * 16];
#pragma unroll
    for (int i = 0; i < 16; i++) dst[i] = src[i];
  }
  __syncthreads();

  float m = -3.0e38f, l = 0.0f;
  v8f acc[8] = {};   // C/D tiles: 8 tiles of 16x16 f32 (V=128)

  const int nn = lane & 15;
  const int g  = lane >> 4;

  for (int c = 0; c < CHUNKS; ++c) {
    const int kb = n0 + c * 32;

    // ---- cooperative V^T chunk stage: 128 rows x 32 keys f16 ----
    {
      const uint4* src = (const uint4*)(Vt + (size_t)tid * N_K + kb);
      uint4* dst = &sm.vt4[tid][0];
#pragma unroll
      for (int j = 0; j < 4; j++) dst[j] = src[j];
    }
    __syncthreads();

    // ---- score pass: lane owns key (kb+lane); packed-f16 L1 distance ----
    const h8* kp = (const h8*)(Kh + (size_t)(kb + lane) * D_DIM);
    {
      h8 kreg[16];
#pragma unroll
      for (int i = 0; i < 16; i++) kreg[i] = kp[i];      // dims 0..127
      for (int q = 0; q < 16; ++q) {
        const h8* qp = &sm.qs[w][q][0];
        h8 a0 = (h8)(_Float16)0.0f, a1 = (h8)(_Float16)0.0f;
#pragma unroll
        for (int i = 0; i < 16; i += 2) {
          a0 += habs8(qp[i] - kreg[i]);
          a1 += habs8(qp[i + 1] - kreg[i + 1]);
        }
        sm.S[w][q][lane] = rsum8(a0 + a1);               // positive partial L1
      }
    }
    {
      h8 kreg[16];
#pragma unroll
      for (int i = 0; i < 16; i++) kreg[i] = kp[16 + i]; // dims 128..255
      for (int q = 0; q < 16; ++q) {
        const h8* qp = &sm.qs[w][q][16];
        h8 a0 = (h8)(_Float16)0.0f, a1 = (h8)(_Float16)0.0f;
#pragma unroll
        for (int i = 0; i < 16; i += 2) {
          a0 += habs8(qp[i] - kreg[i]);
          a1 += habs8(qp[i + 1] - kreg[i + 1]);
        }
        float tot = sm.S[w][q][lane] + rsum8(a0 + a1);
        sm.S[w][q][lane] = -tot;                         // score = -L1
      }
    }
    dswait();

    // ---- transposed read into WMMA-A order + online softmax ----
    // A f16 16x32 layout: lane<16 -> K = {0..7,16..23}; lane>=16 -> K = {8..15,24..31}
    const float* Srow = &sm.S[w][nn][0];
    float4 c0 = *(const float4*)(Srow + g * 8);
    float4 c1 = *(const float4*)(Srow + g * 8 + 4);
    float4 c2 = *(const float4*)(Srow + 16 + g * 8);
    float4 c3 = *(const float4*)(Srow + 16 + g * 8 + 4);
    float p[16] = {c0.x, c0.y, c0.z, c0.w, c1.x, c1.y, c1.z, c1.w,
                   c2.x, c2.y, c2.z, c2.w, c3.x, c3.y, c3.z, c3.w};

    float rm = p[0];
#pragma unroll
    for (int j = 1; j < 16; j++) rm = fmaxf(rm, p[j]);
    rm = fmaxf(rm, xor16f(rm));                 // full 32-key row max
    float mn = fmaxf(m, rm);
    float sc = exp2f((m - mn) * L2E);

    float psum = 0.0f;
#pragma unroll
    for (int j = 0; j < 16; j++) {
      p[j] = exp2f((p[j] - mn) * L2E);
      psum += p[j];
    }
    psum += xor16f(psum);
    l = l * sc + psum;
    m = mn;

    // ---- broadcast per-query rescale factors into C layout ----
    if (lane < 16) sm.scales[w][lane] = sc;
    dswait();
    float4 s01 = *(const float4*)&sm.scales[w][g * 8];
    float4 s23 = *(const float4*)&sm.scales[w][g * 8 + 4];
#pragma unroll
    for (int t = 0; t < 8; t++) {
      acc[t][0] *= s01.x; acc[t][1] *= s01.y;
      acc[t][2] *= s01.z; acc[t][3] *= s01.w;
      acc[t][4] *= s23.x; acc[t][5] *= s23.y;
      acc[t][6] *= s23.z; acc[t][7] *= s23.w;
    }

    // ---- build A operand (probabilities, f16) ----
    v16h A = {(_Float16)p[0],  (_Float16)p[1],  (_Float16)p[2],  (_Float16)p[3],
              (_Float16)p[4],  (_Float16)p[5],  (_Float16)p[6],  (_Float16)p[7],
              (_Float16)p[8],  (_Float16)p[9],  (_Float16)p[10], (_Float16)p[11],
              (_Float16)p[12], (_Float16)p[13], (_Float16)p[14], (_Float16)p[15]};

    // ---- P(16x32) x V(32x16) WMMA, 8 value tiles ----
#pragma unroll
    for (int t = 0; t < 8; t++) {
      const uint4* bp = &sm.vt4[t * 16 + nn][0];  // row n, K=g*8..g*8+7 and 16+g*8..16+g*8+7
      h8 lo = __builtin_bit_cast(h8, bp[g]);
      h8 hi = __builtin_bit_cast(h8, bp[2 + g]);
      v16h Bm = __builtin_shufflevector(lo, hi, 0, 1, 2, 3, 4, 5, 6, 7,
                                        8, 9, 10, 11, 12, 13, 14, 15);
      acc[t] = __builtin_amdgcn_wmma_f32_16x16x32_f16(
          false, A, false, Bm, (short)0, acc[t], false, false);
    }
    __syncthreads();   // protect sm.vt4 before next chunk restage
  }

  // ---- write partial (acc in C layout, plus m and l) ----
  float* pb = part + ((size_t)qt * NSPLIT + split) * PART_STRIDE;
#pragma unroll
  for (int t = 0; t < 8; t++)
#pragma unroll
    for (int r = 0; r < 8; r++)
      pb[(r + 8 * g) * V_DIM + t * 16 + nn] = acc[t][r];
  if (lane < 16) {
    pb[2048 + lane] = m;
    pb[2064 + lane] = l;
  }
}

// ---------- kernel 3: merge N-splits (log-sum-exp combine) ----------
__global__ __launch_bounds__(256) void nd_combine(const float* __restrict__ part,
                                                  float* __restrict__ out) {
  int id = blockIdx.x * 256 + threadIdx.x;    // 0..131071
  int b = id >> 7, vv = id & 127;
  int qt = b >> 4, q = b & 15;
  const float* base = part + (size_t)qt * NSPLIT * PART_STRIDE;

  float mg = -3.0e38f;
#pragma unroll 4
  for (int s = 0; s < NSPLIT; s++)
    mg = fmaxf(mg, base[s * PART_STRIDE + 2048 + q]);

  float num = 0.0f, den = 0.0f;
#pragma unroll 4
  for (int s = 0; s < NSPLIT; s++) {
    float ms = base[s * PART_STRIDE + 2048 + q];
    float ls = base[s * PART_STRIDE + 2064 + q];
    float wgt = exp2f((ms - mg) * L2E);
    num += base[s * PART_STRIDE + q * V_DIM + vv] * wgt;
    den += ls * wgt;
  }
  out[id] = num / den;
}

// ---------- launch ----------
extern "C" void kernel_launch(void* const* d_in, const int* in_sizes, int n_in,
                              void* d_out, int out_size, void* d_ws, size_t ws_size,
                              hipStream_t stream) {
  (void)in_sizes; (void)n_in; (void)out_size; (void)ws_size;
  const float* q = (const float*)d_in[0];
  const float* k = (const float*)d_in[1];
  const float* v = (const float*)d_in[2];

  char* ws = (char*)d_ws;
  _Float16* Qh  = (_Float16*)(ws);                                   // 0.5 MB
  _Float16* Kh  = (_Float16*)(ws + 524288);                          // 16 MB
  _Float16* Vt  = (_Float16*)(ws + 524288 + 16777216);               // 8 MB
  float*    prt = (float*)   (ws + 524288 + 16777216 + 8388608);     // 17 MB

  // stage f16 operands
  nd_cvt_qk<<<(N_K * D_DIM) / 256, 256, 0, stream>>>(q, k, Qh, Kh);
  nd_transpose_v<<<dim3(N_K / 32, V_DIM / 32), dim3(32, 8), 0, stream>>>(v, Vt);

  // main pass: 16 query-blocks (64 queries each) x 32 N-splits
  nd_main<<<dim3(B_Q / 64, NSPLIT), 128, 0, stream>>>(Qh, Kh, Vt, prt);

  // combine splits
  nd_combine<<<(B_Q * V_DIM) / 256, 256, 0, stream>>>(prt, (float*)d_out);
}
